// RelPNHead_67190468378871
// MI455X (gfx1250) — matrix operand order repaired and checked by reading, not compiled, not run on hardware
//
#include <hip/hip_runtime.h>
#include <hip/hip_bf16.h>

#define NN        8192
#define FEATN     255
#define HIDN      64
#define PRE_NMS_N 6000
#define POST_NMS_N 300
#define NMS_TH    0.7f
#define NBINS     4096
#define CAPC      32768

typedef __attribute__((ext_vector_type(16))) __bf16 v16bf;
typedef __attribute__((ext_vector_type(8)))  float  v8f;
typedef __attribute__((ext_vector_type(4)))  unsigned u32x4;
typedef __attribute__((ext_vector_type(8)))  int      i32x8;
typedef __attribute__((ext_vector_type(4)))  int      i32x4;

// ---------------- Tensor Data Mover: build D# and issue (ISA ch.8) -----------
static __device__ __forceinline__ unsigned lds_byte_off(const void* p) {
  return (unsigned)(size_t)p;  // generic ptr to LDS: low 32 bits = LDS byte addr
}

// 2-D tile load Global->LDS. dsize_log: 0=1B,1=2B,2=4B. dims/tiles in elements.
// Rows >= dim1 are zero-filled by hardware (OOB reads return zero).
static __device__ __forceinline__ void tdm_load_2d(const void* gaddr, unsigned lds_off,
                                                   unsigned dsize_log, unsigned dim0,
                                                   unsigned dim1, unsigned tile0,
                                                   unsigned tile1,
                                                   unsigned long long stride0) {
  unsigned long long ga = (unsigned long long)gaddr;
  u32x4 g0;
  g0[0] = 1u;                                               // count=1, user mode
  g0[1] = lds_off;                                          // lds_addr
  g0[2] = (unsigned)(ga & 0xFFFFFFFFull);                   // global_addr[31:0]
  g0[3] = (unsigned)((ga >> 32) & 0x1FFFFFFull) | (2u << 30);  // addr[56:32], type=2
  i32x8 g1;
  g1[0] = (int)(dsize_log << 16);                           // wg_mask=0, data_size
  g1[1] = (int)((dim0 & 0xFFFFu) << 16);                    // abar=0, dim0[15:0]
  g1[2] = (int)(((dim0 >> 16) & 0xFFFFu) | ((dim1 & 0xFFFFu) << 16));
  g1[3] = (int)(((dim1 >> 16) & 0xFFFFu) | (tile0 << 16));  // tile_dim0
  g1[4] = (int)(tile1 & 0xFFFFu);                           // tile_dim1, tile_dim2=0
  g1[5] = (int)(stride0 & 0xFFFFFFFFull);                   // dim0 stride lo
  g1[6] = (int)((stride0 >> 32) & 0xFFFFull);               // stride hi, dim1_stride=0
  g1[7] = 0;
  i32x4 g2 = {1, 0, 0, 0};                                  // tensor_dim2=1, unused
  i32x4 g3 = {0, 0, 0, 0};
#if __clang_major__ >= 23
  i32x8 gz = {};
  __builtin_amdgcn_tensor_load_to_lds(g0, g1, g2, g3, gz, 0);
#else
  __builtin_amdgcn_tensor_load_to_lds(g0, g1, g2, g3, 0);
#endif
}

static __device__ __forceinline__ void tdm_load_1d(const void* gaddr, unsigned lds_off,
                                                   unsigned dsize_log, unsigned n) {
  tdm_load_2d(gaddr, lds_off, dsize_log, n, 1u, n, 0u, (unsigned long long)n);
}

// ---------------- WMMA fragment loaders (wave32, contiguous k = 16*(lane/16) + s) ----
static __device__ __forceinline__ v16bf load_a_f32(const float* s, int ld, int m0,
                                                   int k0, int kmax, int lane) {
  int m  = m0 + (lane & 15);
  int kh = (lane >> 4) & 1;
  v16bf a;
#pragma unroll
  for (int j = 0; j < 8; ++j) {
    int kb = kh * 16 + 2 * j;
    int k  = k0 + kb;
    a[2 * j]     = (k     < kmax) ? (__bf16)s[m * ld + k]     : (__bf16)0.0f;
    a[2 * j + 1] = (k + 1 < kmax) ? (__bf16)s[m * ld + k + 1] : (__bf16)0.0f;
  }
  return a;
}

static __device__ __forceinline__ v16bf load_a_bf16(const __bf16* s, int ld, int m0,
                                                    int k0, int lane) {
  int m  = m0 + (lane & 15);
  int kh = (lane >> 4) & 1;
  v16bf a;
#pragma unroll
  for (int j = 0; j < 8; ++j) {
    int kb = kh * 16 + 2 * j;
    int k  = k0 + kb;
    a[2 * j]     = s[m * ld + k];
    a[2 * j + 1] = s[m * ld + k + 1];
  }
  return a;
}

// B (32x16) from an LDS-resident f32 W-slice (k = 0..31 within slice)
static __device__ __forceinline__ v16bf load_b_lds_f32(const float* s, int ld, int n0,
                                                       int lane) {
  int n  = n0 + (lane & 15);
  int kh = (lane >> 4) & 1;
  v16bf b;
#pragma unroll
  for (int j = 0; j < 8; ++j) {
    int k = kh * 16 + 2 * j;
    b[2 * j]     = (__bf16)s[k * ld + n];
    b[2 * j + 1] = (__bf16)s[(k + 1) * ld + n];
  }
  return b;
}

// B for S = xs * xo^T: B[k][n] = xo[n][k]; per-lane contiguous 32B
static __device__ __forceinline__ v16bf load_bT_bf16(const __bf16* s, int ld, int n0,
                                                     int k0, int lane) {
  int n  = n0 + (lane & 15);
  int kh = (lane >> 4) & 1;
  const __bf16* p = s + n * ld + k0 + kh * 16;
  v16bf b;
#pragma unroll
  for (int j = 0; j < 16; ++j) b[j] = p[j];
  return b;
}

#define WMMA_BF16(A, B, C) \
  __builtin_amdgcn_wmma_f32_16x16x32_bf16(false, (A), false, (B), (short)0, (C), false, false)

// ---------------- Kernel 1: fused 2-layer MLP, fp32 in -> bf16 out -----------
// 64 blocks x 256 threads (8 waves); wave w owns rows [blk*128 + w*16, +16).
// W1 K-slices (32x64 f32) stream through LDS via TDM, double buffered.
__global__ void mlp_kernel(const float* __restrict__ f, const float* __restrict__ W1,
                           const float* __restrict__ b1, const float* __restrict__ W2,
                           const float* __restrict__ b2, __bf16* __restrict__ xout) {
  __shared__ float  lw[2][32 * HIDN];    // 16 KB: W1 slice double buffer
  __shared__ float  lw2[HIDN * HIDN];    // 16 KB: W2
  __shared__ __bf16 lhid[8][16][HIDN];   // 16 KB: relu(hidden) staging
  int lane = threadIdx.x & 31, wave = threadIdx.x >> 5;
  int m0 = (blockIdx.x * 8 + wave) * 16;
  int nlo = lane & 15, mh = lane >> 4;

  if (wave == 0) {
    tdm_load_1d(W2, lds_byte_off(&lw2[0]), 2u, HIDN * HIDN);
    tdm_load_2d(W1, lds_byte_off(&lw[0][0]), 2u, HIDN, FEATN, HIDN, 32u, HIDN);
  }

  v8f acc[4] = {};
  for (int kk = 0; kk < 8; ++kk) {  // K padded 255 -> 256 (TDM zero-fills OOB row)
    if (wave == 0) __builtin_amdgcn_s_wait_tensorcnt(0);
    __syncthreads();  // slice kk visible to all waves
    if (wave == 0 && kk < 7)  // prefetch next slice into the other buffer
      tdm_load_2d(W1 + (kk + 1) * 32 * HIDN, lds_byte_off(&lw[(kk + 1) & 1][0]), 2u,
                  HIDN, (unsigned)(FEATN - (kk + 1) * 32), HIDN, 32u, HIDN);
    const float* Ws = &lw[kk & 1][0];
    v16bf a = load_a_f32(f, FEATN, m0, kk * 32, FEATN, lane);
#pragma unroll
    for (int t = 0; t < 4; ++t) {
      v16bf b = load_b_lds_f32(Ws, HIDN, t * 16, lane);
      acc[t] = WMMA_BF16(a, b, acc[t]);
    }
    __syncthreads();  // all waves done with slice kk before it is overwritten
  }
#pragma unroll
  for (int t = 0; t < 4; ++t)
#pragma unroll
    for (int j = 0; j < 8; ++j) {
      int n = t * 16 + nlo;
      float v = acc[t][j] + b1[n];
      lhid[wave][mh * 8 + j][n] = (__bf16)fmaxf(v, 0.0f);
    }
  __syncthreads();

  v8f acc2[4] = {};
#pragma unroll
  for (int kk = 0; kk < 2; ++kk) {
    v16bf a = load_a_bf16(&lhid[wave][0][0], HIDN, 0, kk * 32, lane);
#pragma unroll
    for (int t = 0; t < 4; ++t) {
      v16bf b = load_b_lds_f32(&lw2[kk * 32 * HIDN], HIDN, t * 16, lane);
      acc2[t] = WMMA_BF16(a, b, acc2[t]);
    }
  }
#pragma unroll
  for (int t = 0; t < 4; ++t)
#pragma unroll
    for (int j = 0; j < 8; ++j) {
      int n = t * 16 + nlo;
      xout[(m0 + mh * 8 + j) * HIDN + n] = (__bf16)(acc2[t][j] + b2[n]);
    }
}

// ---------------- Kernel 2/4: fused pair-score GEMM (never materialize S) ----
// MODE 0: histogram of sigmoid scores.  MODE 1: compact candidates >= *thresh.
// Block covers 128 rows x 64 cols; xs/xo strips staged to LDS via TDM.
template <int MODE>
__global__ void score_kernel(const __bf16* __restrict__ xs, const __bf16* __restrict__ xo,
                             unsigned* __restrict__ hist, const float* __restrict__ thresh,
                             unsigned* __restrict__ counter, float* __restrict__ csc,
                             int* __restrict__ cid) {
  __shared__ unsigned hloc[NBINS];          // 16 KB
  __shared__ __bf16 lxs[128 * HIDN];        // 16 KB
  __shared__ __bf16 lxo[64 * HIDN];         //  8 KB
  int lane = threadIdx.x & 31, wave = threadIdx.x >> 5;
  int m0b = blockIdx.x * 128;
  int n0b = blockIdx.y * 64;
  int nlo = lane & 15, mh = lane >> 4;

  if (wave == 0) {
    tdm_load_1d(xs + (size_t)m0b * HIDN, lds_byte_off(&lxs[0]), 1u, 128u * HIDN);
    tdm_load_1d(xo + (size_t)n0b * HIDN, lds_byte_off(&lxo[0]), 1u, 64u * HIDN);
  }
  if (MODE == 0)
    for (int i = threadIdx.x; i < NBINS; i += blockDim.x) hloc[i] = 0u;
  if (lane == 0)  // speculative prefetch of next column strip (global_prefetch_b8)
    __builtin_prefetch(xo + ((size_t)((n0b + 64) & (NN - 1))) * HIDN, 0, 1);
  float th = (MODE == 1) ? *thresh : 0.0f;
  if (wave == 0) __builtin_amdgcn_s_wait_tensorcnt(0);
  __syncthreads();

  v16bf a0 = load_a_bf16(lxs, HIDN, wave * 16, 0, lane);
  v16bf a1 = load_a_bf16(lxs, HIDN, wave * 16, 32, lane);

  v8f acc[4] = {};
#pragma unroll
  for (int t = 0; t < 4; ++t) {
    v16bf b0 = load_bT_bf16(lxo, HIDN, t * 16, 0, lane);
    v16bf b1 = load_bT_bf16(lxo, HIDN, t * 16, 32, lane);
    acc[t] = WMMA_BF16(a0, b0, acc[t]);
    acc[t] = WMMA_BF16(a1, b1, acc[t]);
  }

#pragma unroll
  for (int t = 0; t < 4; ++t)
#pragma unroll
    for (int j = 0; j < 8; ++j) {
      int m = m0b + wave * 16 + mh * 8 + j;
      int n = n0b + t * 16 + nlo;
      if (m == n) continue;  // reference sets diag to -1 (never in top-k)
      float sc = 1.0f / (1.0f + __expf(-acc[t][j]));
      if (MODE == 0) {
        int bn = (int)(sc * NBINS);
        bn = bn < 0 ? 0 : (bn > NBINS - 1 ? NBINS - 1 : bn);
        atomicAdd(&hloc[bn], 1u);
      } else if (sc >= th) {
        unsigned pos = atomicAdd(counter, 1u);
        if (pos < CAPC) { csc[pos] = sc; cid[pos] = m * NN + n; }
      }
    }

  __syncthreads();
  if (MODE == 0)
    for (int i = threadIdx.x; i < NBINS; i += blockDim.x)
      if (hloc[i]) atomicAdd(&hist[i], hloc[i]);
}

// ---------------- small helper kernels ---------------------------------------
__global__ void init_kernel(unsigned* __restrict__ hist, unsigned* __restrict__ counter) {
  int i = blockIdx.x * blockDim.x + threadIdx.x;
  if (i < NBINS) hist[i] = 0u;
  if (i == 0) *counter = 0u;
}

__global__ void thresh_kernel(const unsigned* __restrict__ hist, float* __restrict__ thresh) {
  if (threadIdx.x != 0 || blockIdx.x != 0) return;
  unsigned cum = 0u;
  int b = NBINS - 1;
  for (; b >= 0; --b) {
    cum += hist[b];
    if (cum >= (unsigned)PRE_NMS_N) break;
  }
  if (b < 0) b = 0;
  *thresh = (float)b / (float)NBINS;  // lower edge of cutoff bin
}

// single-block bitonic sort, descending (score, then idx asc) => deterministic
__global__ void sort_kernel(float* __restrict__ sc, int* __restrict__ id,
                            const unsigned* __restrict__ counter) {
  int cnt = (int)*counter;
  if (cnt > CAPC) cnt = CAPC;
  for (int i = threadIdx.x; i < CAPC; i += blockDim.x)
    if (i >= cnt) { sc[i] = -3.0e38f; id[i] = 0x7FFFFFFF; }
  __syncthreads();
  for (int k = 2; k <= CAPC; k <<= 1) {
    for (int j = k >> 1; j > 0; j >>= 1) {
      for (int i = threadIdx.x; i < CAPC; i += blockDim.x) {
        int ixj = i ^ j;
        if (ixj > i) {
          float s0 = sc[i], s1 = sc[ixj];
          int d0 = id[i], d1 = id[ixj];
          bool before = (s0 > s1) || (s0 == s1 && d0 < d1);
          bool desc = ((i & k) == 0);
          if (desc != before) { sc[i] = s1; sc[ixj] = s0; id[i] = d1; id[ixj] = d0; }
        }
      }
      __syncthreads();
    }
  }
}

__global__ void boxes_kernel(const float* __restrict__ rois, const int* __restrict__ cid,
                             const unsigned* __restrict__ counter, float* __restrict__ ub) {
  int r = blockIdx.x * blockDim.x + threadIdx.x;
  if (r >= PRE_NMS_N) return;
  int cnt = (int)*counter;
  if (cnt > CAPC) cnt = CAPC;
  int idx = (r < cnt) ? cid[r] : 0;
  int sub = idx / NN, obj = idx % NN;
  float sx1 = rois[sub * 5 + 1], sy1 = rois[sub * 5 + 2];
  float sx2 = rois[sub * 5 + 3], sy2 = rois[sub * 5 + 4];
  float ox1 = rois[obj * 5 + 1], oy1 = rois[obj * 5 + 2];
  float ox2 = rois[obj * 5 + 3], oy2 = rois[obj * 5 + 4];
  ub[4 * r + 0] = fminf(sx1, ox1);
  ub[4 * r + 1] = fminf(sy1, oy1);
  ub[4 * r + 2] = fmaxf(sx2, ox2);
  ub[4 * r + 3] = fmaxf(sy2, oy2);
}

// matrix-NMS as in reference: suppressed[j] = any_{i<j} IoU(i,j) > thresh
__global__ void nms_kernel(const float* __restrict__ ub, int* __restrict__ supp) {
  int j = blockIdx.x * blockDim.x + threadIdx.x;
  if (j >= PRE_NMS_N) return;
  float xj1 = ub[4 * j], yj1 = ub[4 * j + 1], xj2 = ub[4 * j + 2], yj2 = ub[4 * j + 3];
  float aj = (xj2 - xj1) * (yj2 - yj1);
  int s = 0;
  for (int i = 0; i < j; ++i) {
    float xi1 = ub[4 * i], yi1 = ub[4 * i + 1], xi2 = ub[4 * i + 2], yi2 = ub[4 * i + 3];
    float iw = fminf(xi2, xj2) - fmaxf(xi1, xj1);
    float ih = fminf(yi2, yj2) - fmaxf(yi1, yj1);
    float inter = fmaxf(iw, 0.0f) * fmaxf(ih, 0.0f);
    float ai = (xi2 - xi1) * (yi2 - yi1);
    float iou = inter / (ai + aj - inter + 1e-8f);
    if (iou > NMS_TH) { s = 1; break; }
  }
  supp[j] = s;
}

// top-300 of kept scores: unsuppressed in sorted order, then suppressed (-1)
__global__ void final_kernel(const float* __restrict__ csc, const int* __restrict__ cid,
                             const int* __restrict__ supp, const unsigned* __restrict__ counter,
                             float* __restrict__ out) {
  if (threadIdx.x != 0 || blockIdx.x != 0) return;
  int cnt = (int)*counter;
  if (cnt > CAPC) cnt = CAPC;
  if (cnt > PRE_NMS_N) cnt = PRE_NMS_N;
  int w = 0;
  for (int r = 0; r < cnt && w < POST_NMS_N; ++r)
    if (!supp[r]) {
      int idx = cid[r];
      out[2 * w] = (float)(idx / NN);
      out[2 * w + 1] = (float)(idx % NN);
      out[2 * POST_NMS_N + w] = csc[r];
      ++w;
    }
  for (int r = 0; r < cnt && w < POST_NMS_N; ++r)
    if (supp[r]) {
      int idx = cid[r];
      out[2 * w] = (float)(idx / NN);
      out[2 * w + 1] = (float)(idx % NN);
      out[2 * POST_NMS_N + w] = -1.0f;
      ++w;
    }
  for (; w < POST_NMS_N; ++w) {
    out[2 * w] = 0.0f;
    out[2 * w + 1] = 0.0f;
    out[2 * POST_NMS_N + w] = -1.0f;
  }
}

// ---------------- host launch -------------------------------------------------
extern "C" void kernel_launch(void* const* d_in, const int* in_sizes, int n_in,
                              void* d_out, int out_size, void* d_ws, size_t ws_size,
                              hipStream_t stream) {
  (void)in_sizes; (void)n_in; (void)out_size; (void)ws_size;
  const float* rois = (const float*)d_in[0];
  const float* feat = (const float*)d_in[1];
  const float* Ws1  = (const float*)d_in[2];
  const float* bs1  = (const float*)d_in[3];
  const float* Ws2  = (const float*)d_in[4];
  const float* bs2  = (const float*)d_in[5];
  const float* Wo1  = (const float*)d_in[6];
  const float* bo1  = (const float*)d_in[7];
  const float* Wo2  = (const float*)d_in[8];
  const float* bo2  = (const float*)d_in[9];

  char* ws = (char*)d_ws;
  size_t off = 0;
  __bf16* xs = (__bf16*)(ws + off); off += (size_t)NN * HIDN * sizeof(__bf16);
  __bf16* xo = (__bf16*)(ws + off); off += (size_t)NN * HIDN * sizeof(__bf16);
  unsigned* hist    = (unsigned*)(ws + off); off += (size_t)NBINS * sizeof(unsigned);
  unsigned* counter = (unsigned*)(ws + off); off += 256;
  float* thresh     = (float*)(ws + off);    off += 256;
  float* csc = (float*)(ws + off); off += (size_t)CAPC * sizeof(float);
  int*   cid = (int*)(ws + off);   off += (size_t)CAPC * sizeof(int);
  float* ub  = (float*)(ws + off); off += (size_t)PRE_NMS_N * 4 * sizeof(float);
  int*   supp = (int*)(ws + off);  off += (size_t)PRE_NMS_N * sizeof(int);
  float* out = (float*)d_out;

  // 1) MLPs (WMMA bf16, TDM-staged weights): x_sub, x_obj = 8192x64 bf16
  mlp_kernel<<<64, 256, 0, stream>>>(feat, Ws1, bs1, Ws2, bs2, xs);
  mlp_kernel<<<64, 256, 0, stream>>>(feat, Wo1, bo1, Wo2, bo2, xo);

  // 2) reset per-call state (graph-replay safe)
  init_kernel<<<(NBINS + 255) / 256, 256, 0, stream>>>(hist, counter);

  // 3) fused score GEMM pass 1: histogram (no 256MB S materialization)
  dim3 g(NN / 128, NN / 64);
  score_kernel<0><<<g, 256, 0, stream>>>(xs, xo, hist, thresh, counter, csc, cid);

  // 4) exact top-PRE_NMS threshold from histogram tail
  thresh_kernel<<<1, 1, 0, stream>>>(hist, thresh);

  // 5) fused score GEMM pass 2: compact candidates >= threshold
  score_kernel<1><<<g, 256, 0, stream>>>(xs, xo, hist, thresh, counter, csc, cid);

  // 6) sort candidates desc, 7) union boxes, 8) NMS, 9) top-300 emit
  sort_kernel<<<1, 1024, 0, stream>>>(csc, cid, counter);
  boxes_kernel<<<(PRE_NMS_N + 255) / 256, 256, 0, stream>>>(rois, cid, counter, ub);
  nms_kernel<<<(PRE_NMS_N + 255) / 256, 256, 0, stream>>>(ub, supp);
  final_kernel<<<1, 1, 0, stream>>>(csc, cid, supp, counter, out);
}
